// ConditionModel_72335839199364
// MI455X (gfx1250) — compile-verified, matrix operand order checked by reading, not run on hardware
//
#include <hip/hip_runtime.h>
#include <hip/hip_bf16.h>

#define B_    8
#define N_    512
#define L_    512
#define E_    65536
#define NNODE 4096         /* B_*N_ */
#define EA_   69632        /* E_ + NNODE (self loops) */
#define H_    16
#define DH_   4

typedef _Float16 f16;
typedef _Float16 v16h __attribute__((ext_vector_type(16)));
typedef float    v8f  __attribute__((ext_vector_type(8)));

__device__ __forceinline__ float d_lrelu(float x){ return x >= 0.f ? x : 0.01f*x; }
__device__ __forceinline__ float d_sigm (float x){ return 1.f/(1.f+expf(-x)); }
__device__ __forceinline__ unsigned fenc(float f){ unsigned u=__float_as_uint(f); return (u&0x80000000u)? ~u : (u|0x80000000u); }
__device__ __forceinline__ float    fdec(unsigned u){ return __uint_as_float((u&0x80000000u)? (u^0x80000000u) : ~u); }

/* ---------------- generic small kernels ---------------- */

__global__ void k_f2h(const float* __restrict__ s, f16* __restrict__ d, long n){
  long i = (long)blockIdx.x*blockDim.x + threadIdx.x;
  if (i < n) d[i] = (f16)s[i];
}

__global__ void k_zero_f32(float* p, int n){
  int i = blockIdx.x*blockDim.x + threadIdx.x;
  if (i < n) p[i] = 0.f;
}

__global__ void k_fill_u32(unsigned* p, unsigned v, int n){
  int i = blockIdx.x*blockDim.x + threadIdx.x;
  if (i < n) p[i] = v;
}

__global__ void k_lrelu_ip(float* p, int n){
  int i = blockIdx.x*blockDim.x + threadIdx.x;
  if (i < n) p[i] = d_lrelu(p[i]);
}

/* scalar GEMM: C[m][n] = act(sum_k A[m][k]*W[n][k] + b[n]) for tiny layers */
__global__ void k_lin_naive(const float* __restrict__ A, const float* __restrict__ W,
                            const float* __restrict__ bias, float* __restrict__ C,
                            int M, int N, int K, int act){
  int t = blockIdx.x*blockDim.x + threadIdx.x;
  if (t >= M*N) return;
  int m = t / N, n = t % N;
  const float* a = A + (size_t)m*K;
  const float* w = W + (size_t)n*K;
  float s = bias ? bias[n] : 0.f;
  for (int k=0;k<K;k++) s += a[k]*w[k];
  if (act==1) s = d_lrelu(s); else if (act==2) s = fmaxf(s,0.f);
  C[t] = s;
}

__global__ void k_scale_exp(float* v, const float* t, int n){
  int i = blockIdx.x*blockDim.x + threadIdx.x;
  if (i < n) v[i] *= expf(t[0]);
}

/* per-graph LayerNorm over N_*C entries; elementwise affine per feature */
__global__ void k_graph_ln(float* __restrict__ x, const float* __restrict__ w,
                           const float* __restrict__ b, int C){
  int g = blockIdx.x;
  float* xg = x + (size_t)g*N_*C;
  int n = N_*C;
  __shared__ float s1[256], s2[256];
  __shared__ float smean, sinv;
  float a=0.f, q=0.f;
  for (int i=threadIdx.x;i<n;i+=blockDim.x){ float v=xg[i]; a+=v; q+=v*v; }
  s1[threadIdx.x]=a; s2[threadIdx.x]=q; __syncthreads();
  for (int st=blockDim.x>>1; st>0; st>>=1){
    if ((int)threadIdx.x < st){ s1[threadIdx.x]+=s1[threadIdx.x+st]; s2[threadIdx.x]+=s2[threadIdx.x+st]; }
    __syncthreads();
  }
  if (threadIdx.x==0){
    float m = s1[0]/n; float v = s2[0]/n - m*m;
    smean = m; sinv = rsqrtf(v + 1e-5f);
  }
  __syncthreads();
  for (int i=threadIdx.x;i<n;i+=blockDim.x){
    int c = i % C;
    xg[i] = (xg[i]-smean)*sinv*w[c] + b[c];
  }
}

/* per-row LayerNorm, block per row */
__global__ void k_row_ln(float* __restrict__ x, const float* __restrict__ w,
                         const float* __restrict__ b, int C){
  float* xr = x + (size_t)blockIdx.x*C;
  __shared__ float s1[256], s2[256];
  __shared__ float smean, sinv;
  float a=0.f, q=0.f;
  for (int i=threadIdx.x;i<C;i+=blockDim.x){ float v=xr[i]; a+=v; q+=v*v; }
  s1[threadIdx.x]=a; s2[threadIdx.x]=q; __syncthreads();
  for (int st=blockDim.x>>1; st>0; st>>=1){
    if ((int)threadIdx.x < st){ s1[threadIdx.x]+=s1[threadIdx.x+st]; s2[threadIdx.x]+=s2[threadIdx.x+st]; }
    __syncthreads();
  }
  if (threadIdx.x==0){
    float m = s1[0]/C; float v = s2[0]/C - m*m;
    smean = m; sinv = rsqrtf(v + 1e-5f);
  }
  __syncthreads();
  for (int i=threadIdx.x;i<C;i+=blockDim.x)
    xr[i] = (xr[i]-smean)*sinv*w[i] + b[i];
}

/* LayerNorm over last dim == 2, thread per row */
__global__ void k_ln2(float* __restrict__ x, const float* __restrict__ w,
                      const float* __restrict__ b, int M){
  int i = blockIdx.x*blockDim.x + threadIdx.x;
  if (i >= M) return;
  float a = x[i*2], c = x[i*2+1];
  float m = 0.5f*(a+c);
  float v = 0.5f*((a-m)*(a-m)+(c-m)*(c-m));
  float inv = rsqrtf(v + 1e-5f);
  x[i*2]   = (a-m)*inv*w[0] + b[0];
  x[i*2+1] = (c-m)*inv*w[1] + b[1];
}

/* ---------------- WMMA GEMM ----------------
 * C(MxN,f32) = act(A(MxK,f16) @ W(NxK,f16)^T + bias) + resid, optional C-accumulate.
 * A row m maps to memory row (rowOff + m*rowStride) — enables even/odd LSTM pair gather.
 * Each wave computes a 32x32 C macro-tile (2x2 of 16x16) reusing A/B fragments.
 * OOB rows/cols are CLAMPED (not zero-filled): garbage lands only in C entries the
 * guarded epilogue never stores, keeping the K-loop branch-free (no exec juggling).
 * Fragment layout per CDNA5 ISA 7.12.2 (16-bit A 16x32): lane l (l<16): rows m0+l,
 * kchunks {k0..k0+7, k0+16..k0+23}; l>=16: kchunks shifted by 8. B^T rows share it. */
__device__ __forceinline__ v16h frag_load(const f16* __restrict__ rowbase, int kb){
  union { v16h v; uint4 q[2]; } u;
  const uint4* p = (const uint4*)(rowbase + kb);   /* 16B aligned: kb%8==0, ld%8==0 */
  u.q[0] = p[0];      /* halves kb .. kb+7   */
  u.q[1] = p[2];      /* halves kb+16..kb+23 */
  return u.v;
}

__global__ void __launch_bounds__(256)
k_wmma_gemm(const f16* __restrict__ A, int lda, int rowOff, int rowStride,
            const f16* __restrict__ Bw,
            float* __restrict__ C, int ldc,
            const float* __restrict__ bias, const float* __restrict__ resid,
            int M, int N, int K, int acc, int act)
{
  const int lane = threadIdx.x & 31;
  const int wave = blockIdx.x*8 + (threadIdx.x>>5);
  const int n2 = (N+31)>>5;
  const int m2 = (M+31)>>5;
  if (wave >= m2*n2) return;
  const int m0 = (wave / n2) << 5;
  const int n0 = (wave % n2) << 5;
  const int half = lane >> 4;
  const int l15  = lane & 15;

  v8f c00, c01, c10, c11;
  for (int r=0;r<8;r++){ c00[r]=0.f; c01[r]=0.f; c10[r]=0.f; c11[r]=0.f; }
  if (acc){
    /* clamped gather: OOB entries pick up garbage that is never stored back */
    for (int r=0;r<8;r++){
      int ma = min(m0 + half*8 + r, M-1), mb = min(m0 + half*8 + r + 16, M-1);
      int na = min(n0 + l15, N-1),        nb = min(n0 + 16 + l15, N-1);
      c00[r]=C[(size_t)ma*ldc+na];
      c01[r]=C[(size_t)ma*ldc+nb];
      c10[r]=C[(size_t)mb*ldc+na];
      c11[r]=C[(size_t)mb*ldc+nb];
    }
  }
  const int mr0 = min(m0 + l15,    M-1);
  const int mr1 = min(m0 + 16+l15, M-1);
  const int nc0 = min(n0 + l15,    N-1);
  const int nc1 = min(n0 + 16+l15, N-1);
  const f16* a0p = A  + ((size_t)rowOff + (size_t)mr0*rowStride)*lda;
  const f16* a1p = A  + ((size_t)rowOff + (size_t)mr1*rowStride)*lda;
  const f16* b0p = Bw + (size_t)nc0*K;
  const f16* b1p = Bw + (size_t)nc1*K;

  for (int k0=0;k0<K;k0+=32){
    int kb = k0 + half*8;
    v16h a0 = frag_load(a0p, kb);
    v16h a1 = frag_load(a1p, kb);
    v16h b0 = frag_load(b0p, kb);
    v16h b1 = frag_load(b1p, kb);
    c00 = __builtin_amdgcn_wmma_f32_16x16x32_f16(false,a0,false,b0,(short)0,c00,false,false);
    c01 = __builtin_amdgcn_wmma_f32_16x16x32_f16(false,a0,false,b1,(short)0,c01,false,false);
    c10 = __builtin_amdgcn_wmma_f32_16x16x32_f16(false,a1,false,b0,(short)0,c10,false,false);
    c11 = __builtin_amdgcn_wmma_f32_16x16x32_f16(false,a1,false,b1,(short)0,c11,false,false);
  }
  for (int r=0;r<8;r++){
    int ma = m0 + half*8 + r, mb = ma+16;
    int na = n0 + l15,        nb = n0+16+l15;
    float v;
    if (ma<M && na<N){ v=c00[r]; if(bias)v+=bias[na]; if(act==1)v=d_lrelu(v); else if(act==2)v=fmaxf(v,0.f); if(resid)v+=resid[(size_t)ma*ldc+na]; C[(size_t)ma*ldc+na]=v; }
    if (ma<M && nb<N){ v=c01[r]; if(bias)v+=bias[nb]; if(act==1)v=d_lrelu(v); else if(act==2)v=fmaxf(v,0.f); if(resid)v+=resid[(size_t)ma*ldc+nb]; C[(size_t)ma*ldc+nb]=v; }
    if (mb<M && na<N){ v=c10[r]; if(bias)v+=bias[na]; if(act==1)v=d_lrelu(v); else if(act==2)v=fmaxf(v,0.f); if(resid)v+=resid[(size_t)mb*ldc+na]; C[(size_t)mb*ldc+na]=v; }
    if (mb<M && nb<N){ v=c11[r]; if(bias)v+=bias[nb]; if(act==1)v=d_lrelu(v); else if(act==2)v=fmaxf(v,0.f); if(resid)v+=resid[(size_t)mb*ldc+nb]; C[(size_t)mb*ldc+nb]=v; }
  }
}

/* ---------------- GATv2 kernels ---------------- */

__global__ void k_gat_loop_accum(const int* __restrict__ src, const int* __restrict__ dst,
                                 const float* __restrict__ e, float* __restrict__ deg,
                                 float* __restrict__ loopsum){
  int i = blockIdx.x*blockDim.x + threadIdx.x;
  if (i >= E_) return;
  (void)src;
  int d = dst[i];
  atomicAdd(&deg[d], 1.f);
  atomicAdd(&loopsum[d*2  ], e[i*2  ]);
  atomicAdd(&loopsum[d*2+1], e[i*2+1]);
}

__global__ void k_gat_loop_fin(const float* __restrict__ loopsum, const float* __restrict__ deg,
                               float* __restrict__ loope){
  int i = blockIdx.x*blockDim.x + threadIdx.x;
  if (i >= NNODE) return;
  float dd = fmaxf(deg[i], 1.f);
  loope[i*2  ] = loopsum[i*2  ]/dd;
  loope[i*2+1] = loopsum[i*2+1]/dd;
}

__global__ void k_gat_logits(const int* __restrict__ src, const int* __restrict__ dst,
                             const float* __restrict__ e, const float* __restrict__ loope,
                             const float* __restrict__ xl, const float* __restrict__ xr,
                             const float* __restrict__ We, const float* __restrict__ att,
                             float* __restrict__ aout, unsigned* __restrict__ amax){
  int i = blockIdx.x*blockDim.x + threadIdx.x;
  if (i >= EA_) return;
  int s, d; float e0, e1;
  if (i < E_){ s = src[i]; d = dst[i]; e0 = e[i*2]; e1 = e[i*2+1]; }
  else       { s = d = i - E_; e0 = loope[s*2]; e1 = loope[s*2+1]; }
  const float* pl = xl + (size_t)s*64;
  const float* pr = xr + (size_t)d*64;
  float a = 0.f;
  for (int c=0;c<64;c++){
    float m = pl[c] + pr[c] + e0*We[c*2] + e1*We[c*2+1];
    a += d_lrelu(m)*att[c];
  }
  aout[i] = a;
  atomicMax(&amax[d], fenc(a));
}

__global__ void k_gat_exp(const int* __restrict__ dst, const float* __restrict__ aout,
                          const unsigned* __restrict__ amax, float* __restrict__ ea,
                          float* __restrict__ easum){
  int i = blockIdx.x*blockDim.x + threadIdx.x;
  if (i >= EA_) return;
  int d = (i < E_) ? dst[i] : (i - E_);
  float w = expf(aout[i] - fdec(amax[d]));
  ea[i] = w;
  atomicAdd(&easum[d], w);
}

__global__ void k_init_bias_rows(float* __restrict__ out, const float* __restrict__ bias){
  int t = blockIdx.x*blockDim.x + threadIdx.x;
  if (t >= NNODE*64) return;
  out[t] = bias[t & 63];
}

__global__ void k_gat_scatter(const int* __restrict__ src, const int* __restrict__ dst,
                              const float* __restrict__ ea, const float* __restrict__ easum,
                              const float* __restrict__ xl, float* __restrict__ out){
  int t = blockIdx.x*blockDim.x + threadIdx.x;
  if (t >= EA_*64) return;
  int i = t >> 6, c = t & 63;
  int s, d;
  if (i < E_){ s = src[i]; d = dst[i]; } else { s = d = i - E_; }
  float alpha = ea[i]/easum[d];
  atomicAdd(&out[(size_t)d*64 + c], alpha * xl[(size_t)s*64 + c]);
}

/* ---------------- cross-attention core (DH=4) ---------------- */

__global__ void __launch_bounds__(512)
k_cross_attn(const float* __restrict__ qh, const float* __restrict__ kh,
             const float* __restrict__ vh, const int* __restrict__ mask,
             float* __restrict__ o){
  __shared__ float kb[L_*DH_];
  __shared__ float vb[L_*DH_];
  __shared__ int   mb[L_];
  int b  = blockIdx.x / H_;
  int hd = blockIdx.x % H_;
  int t  = threadIdx.x;            /* 0..511 : serves as l for load, n for compute */
  const float* kp = kh + (((size_t)b*L_ + t)*H_ + hd)*DH_;
  const float* vp = vh + (((size_t)b*L_ + t)*H_ + hd)*DH_;
  for (int d=0; d<DH_; d++){ kb[t*DH_+d] = kp[d]; vb[t*DH_+d] = vp[d]; }
  mb[t] = mask[b*L_ + t];
  __syncthreads();
  const float* qp = qh + (((size_t)b*N_ + t)*H_ + hd)*DH_;
  float q0=qp[0], q1=qp[1], q2=qp[2], q3=qp[3];
  float mx = -1e30f;
  for (int l=0;l<L_;l++){
    if (mb[l]){
      float s = 0.5f*(q0*kb[l*4] + q1*kb[l*4+1] + q2*kb[l*4+2] + q3*kb[l*4+3]);
      mx = fmaxf(mx, s);
    }
  }
  float sum=0.f, o0=0.f, o1=0.f, o2=0.f, o3=0.f;
  for (int l=0;l<L_;l++){
    if (mb[l]){
      float s = 0.5f*(q0*kb[l*4] + q1*kb[l*4+1] + q2*kb[l*4+2] + q3*kb[l*4+3]);
      float w = expf(s - mx);
      sum += w;
      o0 += w*vb[l*4]; o1 += w*vb[l*4+1]; o2 += w*vb[l*4+2]; o3 += w*vb[l*4+3];
    }
  }
  float inv = 1.f/sum;
  float* op = o + (((size_t)b*N_ + t)*H_ + hd)*DH_;
  op[0]=o0*inv; op[1]=o1*inv; op[2]=o2*inv; op[3]=o3*inv;
}

/* ---------------- LSTM gate kernels (gate order i,f,g,o) ---------------- */

__global__ void k_lstm_step1(const float* __restrict__ G, const float* __restrict__ bih,
                             const float* __restrict__ bhh, float* __restrict__ cOut,
                             f16* __restrict__ hOut, int Mp){
  int t = blockIdx.x*blockDim.x + threadIdx.x;
  if (t >= Mp*1024) return;
  int p = t >> 10, j = t & 1023;
  const float* g = G + (size_t)p*4096;
  float ii = g[j]      + bih[j]      + bhh[j];
  float gg = g[2048+j] + bih[2048+j] + bhh[2048+j];
  float oo = g[3072+j] + bih[3072+j] + bhh[3072+j];
  float c = d_sigm(ii)*tanhf(gg);       /* prev c == 0 */
  float h = d_sigm(oo)*tanhf(c);
  cOut[t] = c;
  hOut[t] = (f16)h;
}

__global__ void k_lstm_step2(const float* __restrict__ G, const float* __restrict__ bih,
                             const float* __restrict__ bhh, const float* __restrict__ cIn,
                             float* __restrict__ tOut, int Mp, int addMode){
  int t = blockIdx.x*blockDim.x + threadIdx.x;
  if (t >= Mp*1024) return;
  int p = t >> 10, j = t & 1023;
  const float* g = G + (size_t)p*4096;   /* G = x2@Wih^T + h1@Whh^T */
  float ii = g[j]      + bih[j]      + bhh[j];
  float ff = g[1024+j] + bih[1024+j] + bhh[1024+j];
  float gg = g[2048+j] + bih[2048+j] + bhh[2048+j];
  float oo = g[3072+j] + bih[3072+j] + bhh[3072+j];
  float c = d_sigm(ff)*cIn[t] + d_sigm(ii)*tanhf(gg);
  float h = d_sigm(oo)*tanhf(c);
  float v = 0.5f*h;
  if (addMode) tOut[t] += v; else tOut[t] = v;
}

/* ================= host orchestration ================= */

static inline long cdivl(long a, long b){ return (a + b - 1)/b; }

extern "C" void kernel_launch(void* const* d_in, const int* in_sizes, int n_in,
                              void* d_out, int out_size, void* d_ws, size_t ws_size,
                              hipStream_t stream) {
  (void)in_sizes; (void)n_in; (void)out_size; (void)ws_size;

  const float* x     = (const float*)d_in[0];
  const int*   eidx  = (const int*)  d_in[1];
  const int*   esrc  = eidx;
  const int*   edst  = eidx + E_;
  const float* efeat = (const float*)d_in[2];
  const float* condp[4] = {(const float*)d_in[4], (const float*)d_in[5],
                           (const float*)d_in[6], (const float*)d_in[7]};
  const int*   amask = (const int*)d_in[8];

  /* params flattened as JAX pytree (dict keys sorted alphabetically), base d_in[9] */
  auto Pf = [&](int i)->const float*{ return (const float*)d_in[9+i]; };
  auto CA = [&](int i, int j)->const float*{ return Pf(16*i + j); };
  /* ca leaf order: 0 d1.b 1 d1.w 2 in_b 3 in_w 4 k.b 5 k.w 6 ln1.b 7 ln1.w
                    8 ln2.b 9 ln2.w 10 out.b 11 out.w 12 q.b 13 q.w 14 v.b 15 v.w */
  auto CONV = [&](int i, int j)->const float*{ return Pf(66 + 7*i + j); };
  /* conv leaf order: 0 We 1 Wl 2 Wr 3 att 4 bias 5 bl 6 br */

  /* workspace layout */
  char* base = (char*)d_ws;
  size_t off = 0;
  auto alloc = [&](size_t bytes)->char*{
    off = (off + 255) & ~(size_t)255;
    char* p = base + off;
    off += bytes;
    return p;
  };

  float* hA     = (float*)alloc((size_t)NNODE*64*4);
  float* hB     = (float*)alloc((size_t)NNODE*64*4);
  float* yb     = (float*)alloc((size_t)NNODE*64*4);
  float* ebuf   = (float*)alloc((size_t)E_*2*4);
  float* xlb    = (float*)alloc((size_t)NNODE*64*4);
  float* xrb    = (float*)alloc((size_t)NNODE*64*4);
  float* deg    = (float*)alloc((size_t)NNODE*4);
  float* loopsum= (float*)alloc((size_t)NNODE*2*4);
  float* loope  = (float*)alloc((size_t)NNODE*2*4);
  unsigned* amax= (unsigned*)alloc((size_t)NNODE*4);
  float* easum  = (float*)alloc((size_t)NNODE*4);
  float* aedge  = (float*)alloc((size_t)EA_*4);
  float* eaedge = (float*)alloc((size_t)EA_*4);
  float* qq     = (float*)alloc((size_t)NNODE*64*4);
  float* kk     = (float*)alloc((size_t)NNODE*64*4);
  float* vv     = (float*)alloc((size_t)NNODE*64*4);
  float* qhB    = (float*)alloc((size_t)NNODE*64*4);
  float* khB    = (float*)alloc((size_t)NNODE*64*4);
  float* vhB    = (float*)alloc((size_t)NNODE*64*4);
  float* obuf   = (float*)alloc((size_t)NNODE*64*4);
  float* tA     = (float*)alloc((size_t)NNODE*1024*4);
  float* tB     = (float*)alloc((size_t)2048*1024*4);
  float* G      = (float*)alloc((size_t)2048*4096*4);
  float* cfb    = (float*)alloc((size_t)2048*1024*4);
  float* cbb    = (float*)alloc((size_t)2048*1024*4);
  float* g1     = (float*)alloc((size_t)8*1024*4);
  float* g2     = (float*)alloc((size_t)8*1024*4);
  f16* h16    = (f16*)alloc((size_t)NNODE*64*2);
  f16* x16    = (f16*)alloc((size_t)NNODE*64*2);
  f16* cond16 = (f16*)alloc((size_t)NNODE*1280*2);
  f16* t16    = (f16*)alloc((size_t)NNODE*1024*2);
  f16* hf16   = (f16*)alloc((size_t)2048*1024*2);
  f16* hb16   = (f16*)alloc((size_t)2048*1024*2);

  auto F2H = [&](const float* s, f16* d, long n){
    k_f2h<<<(unsigned)cdivl(n,256),256,0,stream>>>(s,d,n);
  };
  auto GEMM = [&](const f16* A, int lda, int rowOff, int rowStride, const f16* Bw,
                  float* C, int ldc, const float* bias, const float* resid,
                  int M, int N, int K, int acc, int act){
    long waves = cdivl(M,32)*cdivl(N,32);
    k_wmma_gemm<<<(unsigned)cdivl(waves,8),256,0,stream>>>(
        A,lda,rowOff,rowStride,Bw,C,ldc,bias,resid,M,N,K,acc,act);
  };

  /* convert all WMMA weights to f16 once */
  f16 *caQW[4], *caKW[4], *caVW[4], *caInW[4], *caOutW[4], *caD1W[4];
  for (int i=0;i<4;i++){
    caQW[i]  = (f16*)alloc(64*64*2);    F2H(CA(i,13), caQW[i],  64*64);
    caKW[i]  = (f16*)alloc(64*1280*2);  F2H(CA(i,5),  caKW[i],  64*1280);
    caVW[i]  = (f16*)alloc(64*1280*2);  F2H(CA(i,15), caVW[i],  64*1280);
    caInW[i] = (f16*)alloc(192*64*2);   F2H(CA(i,3),  caInW[i], 192*64);
    caOutW[i]= (f16*)alloc(64*64*2);    F2H(CA(i,11), caOutW[i],64*64);
    caD1W[i] = (f16*)alloc(64*64*2);    F2H(CA(i,1),  caD1W[i], 64*64);
  }
  f16 *convWl16[5] = {0,0,0,0,0}, *convWr16[5] = {0,0,0,0,0};
  for (int i=1;i<5;i++){
    convWl16[i] = (f16*)alloc(64*64*2); F2H(CONV(i,1), convWl16[i], 64*64);
    convWr16[i] = (f16*)alloc(64*64*2); F2H(CONV(i,2), convWr16[i], 64*64);
  }
  f16* lcmLinW = (f16*)alloc((size_t)1024*64*2);      F2H(Pf(126), lcmLinW, 1024*64);
  f16* WihF16  = (f16*)alloc((size_t)4096*1024*2);    F2H(Pf(122), WihF16, (long)4096*1024);
  f16* WhhF16  = (f16*)alloc((size_t)4096*1024*2);    F2H(Pf(121), WhhF16, (long)4096*1024);
  f16* WihB16  = (f16*)alloc((size_t)4096*1024*2);    F2H(Pf(118), WihB16, (long)4096*1024);
  f16* WhhB16  = (f16*)alloc((size_t)4096*1024*2);    F2H(Pf(117), WhhB16, (long)4096*1024);
  f16* d1W16   = (f16*)alloc((size_t)1024*1024*2);    F2H(Pf(102), d1W16, (long)1024*1024);
  f16* d2W16   = (f16*)alloc((size_t)1024*1024*2);    F2H(Pf(104), d2W16, (long)1024*1024);

  /* ---- node/edge embeddings ---- */
  k_lin_naive<<<(unsigned)cdivl(NNODE*16,256),256,0,stream>>>(x, Pf(143), Pf(142), hA, NNODE, 16, 15, 1);
  k_graph_ln<<<B_,256,0,stream>>>(hA, Pf(145), Pf(144), 16);
  k_lin_naive<<<(unsigned)cdivl((long)E_*2,256),256,0,stream>>>(efeat, Pf(106), Pf(105), ebuf, E_, 2, 5, 1);
  k_ln2<<<(unsigned)cdivl(E_,256),256,0,stream>>>(ebuf, Pf(108), Pf(107), E_);

  auto doConv = [&](int i, int cin, float* hin, float* hout){
    k_zero_f32<<<(unsigned)cdivl(NNODE,256),256,0,stream>>>(deg, NNODE);
    k_zero_f32<<<(unsigned)cdivl(NNODE*2,256),256,0,stream>>>(loopsum, NNODE*2);
    k_gat_loop_accum<<<(unsigned)cdivl(E_,256),256,0,stream>>>(esrc, edst, ebuf, deg, loopsum);
    k_gat_loop_fin<<<(unsigned)cdivl(NNODE,256),256,0,stream>>>(loopsum, deg, loope);
    if (cin == 16){
      k_lin_naive<<<(unsigned)cdivl(NNODE*64,256),256,0,stream>>>(hin, CONV(i,1), CONV(i,5), xlb, NNODE,64,16,0);
      k_lin_naive<<<(unsigned)cdivl(NNODE*64,256),256,0,stream>>>(hin, CONV(i,2), CONV(i,6), xrb, NNODE,64,16,0);
    } else {
      F2H(hin, h16, (long)NNODE*64);
      GEMM(h16,64,0,1, convWl16[i], xlb,64, CONV(i,5), nullptr, NNODE,64,64, 0,0);
      GEMM(h16,64,0,1, convWr16[i], xrb,64, CONV(i,6), nullptr, NNODE,64,64, 0,0);
    }
    k_fill_u32<<<(unsigned)cdivl(NNODE,256),256,0,stream>>>(amax, 0x007FFFFFu, NNODE); /* enc(-inf) */
    k_zero_f32<<<(unsigned)cdivl(NNODE,256),256,0,stream>>>(easum, NNODE);
    k_gat_logits<<<(unsigned)cdivl(EA_,256),256,0,stream>>>(esrc,edst,ebuf,loope,xlb,xrb,CONV(i,0),CONV(i,3),aedge,amax);
    k_gat_exp<<<(unsigned)cdivl(EA_,256),256,0,stream>>>(edst,aedge,amax,eaedge,easum);
    k_init_bias_rows<<<(unsigned)cdivl(NNODE*64,256),256,0,stream>>>(hout, CONV(i,4));
    k_gat_scatter<<<(unsigned)cdivl((long)EA_*64,256),256,0,stream>>>(esrc,edst,eaedge,easum,xlb,hout);
    k_lrelu_ip<<<(unsigned)cdivl(NNODE*64,256),256,0,stream>>>(hout, NNODE*64);
    k_graph_ln<<<B_,256,0,stream>>>(hout, Pf(132+2*i), Pf(131+2*i), 64);
  };

  auto doCross = [&](int i, float* hin, float* hout){
    F2H(hin, h16, (long)NNODE*64);
    GEMM(h16,64,0,1, caQW[i], qq,64, CA(i,12), nullptr, NNODE,64,64, 0,0);
    F2H(condp[i], cond16, (long)NNODE*1280);
    GEMM(cond16,1280,0,1, caKW[i], kk,64, CA(i,4),  nullptr, NNODE,64,1280, 0,0);
    GEMM(cond16,1280,0,1, caVW[i], vv,64, CA(i,14), nullptr, NNODE,64,1280, 0,0);
    F2H(qq, x16, (long)NNODE*64);
    GEMM(x16,64,0,1, caInW[i],          qhB,64, CA(i,2),     nullptr, NNODE,64,64, 0,0);
    F2H(kk, x16, (long)NNODE*64);
    GEMM(x16,64,0,1, caInW[i] + 64*64,  khB,64, CA(i,2)+64,  nullptr, NNODE,64,64, 0,0);
    F2H(vv, x16, (long)NNODE*64);
    GEMM(x16,64,0,1, caInW[i] + 128*64, vhB,64, CA(i,2)+128, nullptr, NNODE,64,64, 0,0);
    k_cross_attn<<<B_*H_,512,0,stream>>>(qhB,khB,vhB,amask,obuf);
    F2H(obuf, x16, (long)NNODE*64);
    GEMM(x16,64,0,1, caOutW[i], yb,64, CA(i,10), hin, NNODE,64,64, 0,0);   /* + residual */
    k_row_ln<<<NNODE,64,0,stream>>>(yb, CA(i,7), CA(i,6), 64);
    F2H(yb, x16, (long)NNODE*64);
    GEMM(x16,64,0,1, caD1W[i], hout,64, CA(i,0), yb, NNODE,64,64, 0,1);    /* lrelu then +y */
    k_row_ln<<<NNODE,64,0,stream>>>(hout, CA(i,9), CA(i,8), 64);
  };

  float* hc = hA; float* hn = hB;
  for (int i=0;i<5;i++){
    doConv(i, i==0?16:64, hc, hn);
    { float* t = hc; hc = hn; hn = t; }
    if (i < 4){
      doCross(i, hc, hn);
      { float* t = hc; hc = hn; hn = t; }
    }
  }

  /* ---- LCM pooling ---- */
  F2H(hc, h16, (long)NNODE*64);
  GEMM(h16,64,0,1, lcmLinW, tA,1024, Pf(125), nullptr, NNODE,1024,64, 0,2); /* relu */
  F2H(tA, t16, (long)NNODE*1024);
  float* cur = tA; float* nxt = tB;
  int rows = NNODE;
  const float *bihF=Pf(124), *bhhF=Pf(123), *bihB=Pf(120), *bhhB=Pf(119);
  for (int lv=0; lv<9; lv++){
    int Mp = rows/2;
    long nel = (long)Mp*1024;
    /* forward LSTM over (even, odd) */
    GEMM(t16,1024,0,2, WihF16, G,4096, nullptr,nullptr, Mp,4096,1024, 0,0);
    k_lstm_step1<<<(unsigned)cdivl(nel,256),256,0,stream>>>(G, bihF, bhhF, cfb, hf16, Mp);
    GEMM(t16,1024,1,2, WihF16, G,4096, nullptr,nullptr, Mp,4096,1024, 0,0);
    GEMM(hf16,1024,0,1, WhhF16, G,4096, nullptr,nullptr, Mp,4096,1024, 1,0); /* G += h1@Whh^T */
    k_lstm_step2<<<(unsigned)cdivl(nel,256),256,0,stream>>>(G, bihF, bhhF, cfb, nxt, Mp, 0);
    /* backward LSTM over (odd, even) */
    GEMM(t16,1024,1,2, WihB16, G,4096, nullptr,nullptr, Mp,4096,1024, 0,0);
    k_lstm_step1<<<(unsigned)cdivl(nel,256),256,0,stream>>>(G, bihB, bhhB, cbb, hb16, Mp);
    GEMM(t16,1024,0,2, WihB16, G,4096, nullptr,nullptr, Mp,4096,1024, 0,0);
    GEMM(hb16,1024,0,1, WhhB16, G,4096, nullptr,nullptr, Mp,4096,1024, 1,0);
    k_lstm_step2<<<(unsigned)cdivl(nel,256),256,0,stream>>>(G, bihB, bhhB, cbb, nxt, Mp, 1);
    F2H(nxt, t16, nel);
    { float* t = cur; cur = nxt; nxt = t; }
    rows = Mp;
  }

  /* ---- dense heads ---- */
  GEMM(t16,1024,0,1, d1W16, g1,1024, Pf(101), nullptr, 8,1024,1024, 0,1);
  k_row_ln<<<8,256,0,stream>>>(g1, Pf(128), Pf(127), 1024);
  F2H(g1, x16, 8*1024);
  GEMM(x16,1024,0,1, d2W16, g2,1024, Pf(103), nullptr, 8,1024,1024, 0,1);
  k_row_ln<<<8,256,0,stream>>>(g2, Pf(130), Pf(129), 1024);

  float* out = (float*)d_out;
  k_lin_naive<<<1,32,0,stream>>>(g2, Pf(116), Pf(115), out+0,  8,1,1024,0); /* ki   */
  k_lin_naive<<<1,32,0,stream>>>(g2, Pf(112), Pf(111), out+8,  8,1,1024,0); /* ic50 */
  k_lin_naive<<<1,32,0,stream>>>(g2, Pf(114), Pf(113), out+16, 8,1,1024,0); /* kd   */
  k_lin_naive<<<1,32,0,stream>>>(g2, Pf(110), Pf(109), out+24, 8,1,1024,0); /* ec50 */
  k_lin_naive<<<1,32,0,stream>>>(g2, Pf(65),  Pf(64),  out+32, 8,1,1024,0); /* cls  */
  k_scale_exp<<<1,8,0,stream>>>(out+32, Pf(141), 8);
}
